// Head_43001212567891
// MI455X (gfx1250) — matrix operand order repaired and checked by reading, not compiled, not run on hardware
//
#include <hip/hip_runtime.h>

// Causal single-head attention, B=8 T=2048 N=1024 H=64.
// proj_one: x@W (one of Wk/Wq/Wv per block) via bf16 WMMA.
// attn_head: flash-attention per 16-query tile, bf16 WMMA + online softmax.

#define BATCH 8
#define SEQ   2048
#define NEMB  1024
#define HD    64

typedef __attribute__((ext_vector_type(16))) __bf16 v16bf;
typedef __attribute__((ext_vector_type(8)))  __bf16 v8bf;
typedef __attribute__((ext_vector_type(2)))  __bf16 v2bf;
typedef __attribute__((ext_vector_type(8)))  float  v8f;
typedef __attribute__((ext_vector_type(2)))  float  f2;

// Pair conversion: fptrunc <2 x float> -> <2 x bfloat>; backend selects
// v_cvt_pk_bf16_f32 where available.
__device__ __forceinline__ v2bf cvt2(float lo, float hi) {
  f2 v; v[0] = lo; v[1] = hi;
  return __builtin_convertvector(v, v2bf);
}

// ---------------------------------------------------------------------------
// Projection: y = x @ W for ONE W per block (blockIdx.x/256 selects matrix).
// One wave per 16-row tile of x; 4 accumulators -> low VGPR pressure so the
// four LDS B-fragment loads can issue ahead of the 4-WMMA group.
// ---------------------------------------------------------------------------
__global__ __launch_bounds__(128) void proj_one(
    const float* __restrict__ x,  const float* __restrict__ Wk,
    const float* __restrict__ Wq, const float* __restrict__ Wv,
    __bf16* __restrict__ kbuf,   // [B*T][HD]
    __bf16* __restrict__ qbuf,   // [B*T][HD]
    __bf16* __restrict__ vbufT)  // [B][HD][T] (transposed)
{
  __shared__ alignas(64) __bf16 wlds[HD][32];  // [h][k] 4 KB, bf16-transposed W
  const int tid  = threadIdx.x;
  const int wave = tid >> 5, lane = tid & 31;
  const int m    = lane & 15, half = lane >> 4;
  const int mat  = blockIdx.x >> 8;                       // 0:Wk 1:Wq 2:Wv
  const int row0 = (((blockIdx.x & 255) * 4) + wave) * 16;
  const float* __restrict__ W = (mat == 0) ? Wk : (mat == 1 ? Wq : Wv);

  v8f acc[4];
#pragma unroll
  for (int i = 0; i < 4; ++i) acc[i] = (v8f)0.0f;

  for (int ks = 0; ks < NEMB / 32; ++ks) {
    const int kbase = ks * 32;
    __syncthreads();
    // Stage W chunk as bf16 pairs: wlds[h][2t..2t+1] = W[kbase+2t..][h].
    for (int p = tid; p < 64 * 16; p += 128) {
      const int t = p >> 6;            // k-pair index 0..15
      const int h = p & 63;            // coalesced in h across threads
      *(v2bf*)&wlds[h][2 * t] = cvt2(W[(size_t)(kbase + 2 * t) * HD + h],
                                     W[(size_t)(kbase + 2 * t + 1) * HD + h]);
    }
    __syncthreads();

    // A fragment: row = row0+m, K = 16*half + e (+kbase)
    const float* xp = x + (size_t)(row0 + m) * NEMB + kbase + 16 * half;
    v16bf a;
#pragma unroll
    for (int j = 0; j < 4; ++j) {
      const v2bf p0 = cvt2(xp[2 * j], xp[2 * j + 1]);
      const v2bf p1 = cvt2(xp[8 + 2 * j], xp[9 + 2 * j]);
      a[2 * j]      = p0[0]; a[2 * j + 1]  = p0[1];
      a[8 + 2 * j]  = p1[0]; a[9 + 2 * j]  = p1[1];
    }

    // Four h-tile B fragments, then four WMMAs.
    const v16bf b0 = *(const v16bf*)&wlds[0 * 16 + m][16 * half];
    const v16bf b1 = *(const v16bf*)&wlds[1 * 16 + m][16 * half];
    const v16bf b2 = *(const v16bf*)&wlds[2 * 16 + m][16 * half];
    const v16bf b3 = *(const v16bf*)&wlds[3 * 16 + m][16 * half];
    acc[0] = __builtin_amdgcn_wmma_f32_16x16x32_bf16(false, a, false, b0,
                                                     (short)0, acc[0], false, false);
    acc[1] = __builtin_amdgcn_wmma_f32_16x16x32_bf16(false, a, false, b1,
                                                     (short)0, acc[1], false, false);
    acc[2] = __builtin_amdgcn_wmma_f32_16x16x32_bf16(false, a, false, b2,
                                                     (short)0, acc[2], false, false);
    acc[3] = __builtin_amdgcn_wmma_f32_16x16x32_bf16(false, a, false, b3,
                                                     (short)0, acc[3], false, false);
  }

  if (mat < 2) {
    // k / q: row-major bf16 stores.
    __bf16* dst = (mat == 0) ? kbuf : qbuf;
#pragma unroll
    for (int ht = 0; ht < 4; ++ht)
#pragma unroll
      for (int r = 0; r < 8; ++r)
        dst[(size_t)(row0 + r + 8 * half) * HD + ht * 16 + m] =
            (__bf16)acc[ht][r];
  } else {
    // v transposed: vT[b][h][t]; r contiguous in t -> one b128 per ht.
    const int bb = row0 >> 11, t0 = row0 & (SEQ - 1);
#pragma unroll
    for (int ht = 0; ht < 4; ++ht) {
      v8bf pk;
#pragma unroll
      for (int j = 0; j < 4; ++j) {
        const v2bf p = cvt2(acc[ht][2 * j], acc[ht][2 * j + 1]);
        pk[2 * j] = p[0]; pk[2 * j + 1] = p[1];
      }
      *(v8bf*)&vbufT[((size_t)bb * HD + ht * 16 + m) * SEQ + t0 + 8 * half] = pk;
    }
  }
}

// ---------------------------------------------------------------------------
// Flash attention: one wave per (batch, 16-query tile); 32-key blocks.
// ---------------------------------------------------------------------------
__global__ __launch_bounds__(128) void attn_head(
    const __bf16* __restrict__ qbuf, const __bf16* __restrict__ kbuf,
    const __bf16* __restrict__ vbufT, float* __restrict__ out)
{
  __shared__ alignas(16) __bf16 plds[4][16][40];  // per-wave P tile, padded
  const int tid  = threadIdx.x;
  const int wave = tid >> 5, lane = tid & 31;
  const int m    = lane & 15, half = lane >> 4;
  const int b    = blockIdx.y;
  const int q0   = (blockIdx.x * 4 + wave) * 16;

  // Q A-fragments: rows q0+m, two K-steps over H=64.
  v16bf aq[2];
#pragma unroll
  for (int ks = 0; ks < 2; ++ks) {
    const __bf16* qp = qbuf + (size_t)(b * SEQ + q0 + m) * HD + 32 * ks + 16 * half;
    const v8bf lo = *(const v8bf*)qp;
    const v8bf hi = *(const v8bf*)(qp + 8);
#pragma unroll
    for (int e = 0; e < 8; ++e) { aq[ks][e] = lo[e]; aq[ks][8 + e] = hi[e]; }
  }

  float mr[8], lr[8];
#pragma unroll
  for (int r = 0; r < 8; ++r) { mr[r] = -3.0e38f; lr[r] = 0.0f; }
  v8f o[4];
#pragma unroll
  for (int ht = 0; ht < 4; ++ht) o[ht] = (v8f)0.0f;

  const float sc = 0.125f * 1.44269504088896f;  // H^-0.5 * log2(e)
  const int  nkb = (q0 + 16 + 31) >> 5;

  for (int kbi = 0; kbi < nkb; ++kbi) {
    const int kbase = kbi * 32;

    const __bf16* kp0 = kbuf + (size_t)(b * SEQ + kbase + m) * HD + 16 * half;
    const __bf16* kp1 = kbuf + (size_t)(b * SEQ + kbase + 16 + m) * HD + 16 * half;
    const __bf16* vp  = vbufT + ((size_t)b * HD + m) * SEQ + kbase + 16 * half;

    if (kbi + 1 < nkb) {                       // prefetch next key block
      __builtin_prefetch(kp0 + 32 * HD, 0, 1);
      __builtin_prefetch(kp1 + 32 * HD, 0, 1);
      __builtin_prefetch(vp + 32, 0, 1);
    }

    // S = Q @ K^T for 32 keys: four K fragments, then 4 WMMAs.
    const v16bf b00 = *(const v16bf*)(kp0);
    const v16bf b01 = *(const v16bf*)(kp0 + 32);
    const v16bf b10 = *(const v16bf*)(kp1);
    const v16bf b11 = *(const v16bf*)(kp1 + 32);
    v8f c0 = (v8f)0.0f, c1 = (v8f)0.0f;
    c0 = __builtin_amdgcn_wmma_f32_16x16x32_bf16(false, aq[0], false, b00,
                                                 (short)0, c0, false, false);
    c0 = __builtin_amdgcn_wmma_f32_16x16x32_bf16(false, aq[1], false, b01,
                                                 (short)0, c0, false, false);
    c1 = __builtin_amdgcn_wmma_f32_16x16x32_bf16(false, aq[0], false, b10,
                                                 (short)0, c1, false, false);
    c1 = __builtin_amdgcn_wmma_f32_16x16x32_bf16(false, aq[1], false, b11,
                                                 (short)0, c1, false, false);

    // Online softmax (base-2 domain); row stats live in one 16-lane half.
    float p0[8], p1[8], al[8];
#pragma unroll
    for (int r = 0; r < 8; ++r) {
      const int row = q0 + r + 8 * half;
      float s0 = c0[r] * sc; if (kbase + m      > row) s0 = -1.0e30f;
      float s1 = c1[r] * sc; if (kbase + 16 + m > row) s1 = -1.0e30f;
      float rm = fmaxf(s0, s1);
#pragma unroll
      for (int d = 1; d < 16; d <<= 1) rm = fmaxf(rm, __shfl_xor(rm, d, 32));
      const float mn = fmaxf(mr[r], rm);
      al[r] = __builtin_amdgcn_exp2f(mr[r] - mn);
      p0[r] = __builtin_amdgcn_exp2f(s0 - mn);
      p1[r] = __builtin_amdgcn_exp2f(s1 - mn);
      float rs = p0[r] + p1[r];
#pragma unroll
      for (int d = 1; d < 16; d <<= 1) rs += __shfl_xor(rs, d, 32);
      lr[r] = lr[r] * al[r] + rs;
      mr[r] = mn;
    }
#pragma unroll
    for (int ht = 0; ht < 4; ++ht)
#pragma unroll
      for (int r = 0; r < 8; ++r) o[ht][r] *= al[r];

    // C-layout P -> LDS (row-major 16x32), then reload as A fragment.
    // Same-wave DS ops are in-order, so no barrier needed.
#pragma unroll
    for (int r = 0; r < 8; ++r) {
      const v2bf p = cvt2(p0[r], p1[r]);
      plds[wave][r + 8 * half][m]      = p[0];
      plds[wave][r + 8 * half][16 + m] = p[1];
    }
    const v8bf lo = *(const v8bf*)&plds[wave][m][16 * half];
    const v8bf hi = *(const v8bf*)&plds[wave][m][16 * half + 8];
    v16bf ap;
#pragma unroll
    for (int e = 0; e < 8; ++e) { ap[e] = lo[e]; ap[8 + e] = hi[e]; }

    // O += P @ V : four independent V fragments, then 4 WMMAs.
    const v16bf v0 = *(const v16bf*)(vp + (size_t)(0 * 16) * SEQ);
    const v16bf v1 = *(const v16bf*)(vp + (size_t)(1 * 16) * SEQ);
    const v16bf v2 = *(const v16bf*)(vp + (size_t)(2 * 16) * SEQ);
    const v16bf v3 = *(const v16bf*)(vp + (size_t)(3 * 16) * SEQ);
    o[0] = __builtin_amdgcn_wmma_f32_16x16x32_bf16(false, ap, false, v0,
                                                   (short)0, o[0], false, false);
    o[1] = __builtin_amdgcn_wmma_f32_16x16x32_bf16(false, ap, false, v1,
                                                   (short)0, o[1], false, false);
    o[2] = __builtin_amdgcn_wmma_f32_16x16x32_bf16(false, ap, false, v2,
                                                   (short)0, o[2], false, false);
    o[3] = __builtin_amdgcn_wmma_f32_16x16x32_bf16(false, ap, false, v3,
                                                   (short)0, o[3], false, false);
  }

  // out = O / l  (fp32)
#pragma unroll
  for (int ht = 0; ht < 4; ++ht)
#pragma unroll
    for (int r = 0; r < 8; ++r) {
      const int row = q0 + r + 8 * half;
      out[(size_t)(b * SEQ + row) * HD + ht * 16 + m] = o[ht][r] / lr[r];
    }
}

// ---------------------------------------------------------------------------
extern "C" void kernel_launch(void* const* d_in, const int* in_sizes, int n_in,
                              void* d_out, int out_size, void* d_ws, size_t ws_size,
                              hipStream_t stream) {
  const float* x  = (const float*)d_in[0];
  const float* Wk = (const float*)d_in[1];
  const float* Wq = (const float*)d_in[2];
  const float* Wv = (const float*)d_in[3];
  float* out = (float*)d_out;

  const size_t n = (size_t)BATCH * SEQ * HD;  // 1M elems per tensor
  __bf16* kbuf  = (__bf16*)d_ws;              // 2 MB
  __bf16* qbuf  = kbuf + n;                   // 2 MB
  __bf16* vbufT = qbuf + n;                   // 2 MB

  // 3 matrices x 256 tile-blocks; blockIdx.x>>8 selects Wk/Wq/Wv.
  proj_one<<<dim3(3 * BATCH * SEQ / 64), 128, 0, stream>>>(x, Wk, Wq, Wv,
                                                           kbuf, qbuf, vbufT);
  attn_head<<<dim3(SEQ / 64, BATCH), 128, 0, stream>>>(qbuf, kbuf, vbufT, out);
}
